// TVMinimization_56573309224254
// MI455X (gfx1250) — matrix operand order repaired and checked by reading, not compiled, not run on hardware
//
#include <hip/hip_runtime.h>
#include <stdint.h>

typedef __attribute__((ext_vector_type(16))) _Float16 v16h;
typedef __attribute__((ext_vector_type(8)))  float    v8f;

#define BB   8
#define HH   128
#define WW   128
#define NN   16384   // H*W
#define MM   4096
#define TAU  0.01f
#define EPSV 1e-8f
#define NITER 100
#define KS1  16      // GEMM1 split-K slices (512 chunks / 16 = 32 each)
#define KS2  2       // GEMM2 split-K slices (128 chunks / 2 = 64 each)

// ---------------------------------------------------------------------------
// One-time conversion: A (fp32, MxN row-major) -> A1 (f16, WMMA A-operand
// tiles: m-tile of 16 rows x n-chunk of 32 K). Block = 1KB; lane's 32B
// contiguous at lane*32.
// K pair start this packing uses for (lane l, vgpr j):
//   k0 = 2j + (l>=16 ? 16 : 0)
// ---------------------------------------------------------------------------
__global__ void k_conv_a1(const float* __restrict__ A, _Float16* __restrict__ A1) {
  uint32_t tid = blockIdx.x * blockDim.x + threadIdx.x;   // 2^25 threads
  uint32_t j  = tid & 7u;
  uint32_t l  = (tid >> 3) & 31u;
  uint32_t c  = (tid >> 8) & 511u;     // n-chunk (N/32 = 512)
  uint32_t mt = tid >> 17;             // m-tile  (M/16 = 256)
  uint32_t m  = mt * 16u + (l & 15u);
  uint32_t k0 = 2u * j + ((l >= 16u) ? 16u : 0u);
  size_t n = (size_t)c * 32u + k0;
  const float* src = A + (size_t)m * NN + n;
  _Float16* dst = A1 + ((size_t)(mt * 512u + c) * 512u) + (size_t)l * 16u + (size_t)j * 2u;
  dst[0] = (_Float16)src[0];
  dst[1] = (_Float16)src[1];
}

// ---------------------------------------------------------------------------
// One-time conversion: A -> A2 (f16, WMMA B-operand tiles: m-chunk of 32 K x
// n-tile of 16 cols).  K pair start used here (same as A1): k0 = 2j + (l>=16 ? 16 : 0).
// ---------------------------------------------------------------------------
__global__ void k_conv_a2(const float* __restrict__ A, _Float16* __restrict__ A2) {
  uint32_t tid = blockIdx.x * blockDim.x + threadIdx.x;   // 2^25 threads
  uint32_t j  = tid & 7u;
  uint32_t l  = (tid >> 3) & 31u;
  uint32_t c  = (tid >> 8) & 127u;     // m-chunk (M/32 = 128)
  uint32_t nt = tid >> 15;             // n-tile  (N/16 = 1024)
  uint32_t n  = nt * 16u + (l & 15u);
  uint32_t k0 = 2u * j + ((l >= 16u) ? 16u : 0u);
  uint32_t m  = c * 32u + k0;
  _Float16* dst = A2 + ((size_t)(nt * 128u + c) * 512u) + (size_t)l * 16u + (size_t)j * 2u;
  dst[0] = (_Float16)A[(size_t)m * NN + n];
  dst[1] = (_Float16)A[(size_t)(m + 1u) * NN + n];
}

__global__ void k_zero(float* __restrict__ p, int n) {
  int i = blockIdx.x * blockDim.x + threadIdx.x;
  if (i < n) p[i] = 0.0f;
}

// ---------------------------------------------------------------------------
// Pointwise: v = x - TAU*div(y_cur);  y_next = normalize(y_cur + TAU*grad(xbar))
// ---------------------------------------------------------------------------
__global__ void k_pointwise(const float* __restrict__ x, const float* __restrict__ xbar,
                            const float* __restrict__ ycur, float* __restrict__ ynext,
                            float* __restrict__ v) {
  uint32_t tid = blockIdx.x * blockDim.x + threadIdx.x;   // B*N = 131072
  uint32_t b   = tid >> 14;
  uint32_t rem = tid & 16383u;
  uint32_t h = rem >> 7, w = rem & 127u;
  const float* xb  = x    + (size_t)b * NN;
  const float* xbb = xbar + (size_t)b * NN;
  const float* y0  = ycur + (size_t)b * 2u * NN;
  const float* y1  = y0 + NN;
  float px = y0[rem], py = y1[rem];
  float dx = (w > 0u) ? (px - y0[rem - 1u])  : 0.0f;
  float dy = (h > 0u) ? (py - y1[rem - WW])  : 0.0f;
  v[(size_t)b * NN + rem] = xb[rem] - TAU * (dx + dy);
  float xc = xbb[rem];
  float gx = (w < WW - 1u) ? (xc - xbb[rem + 1u]) : 0.0f;
  float gy = (h < HH - 1u) ? (xc - xbb[rem + WW]) : 0.0f;
  float yx = px + TAU * gx;
  float yy = py + TAU * gy;
  float d  = fmaxf(fabsf(yx), fabsf(yy)) + EPSV;
  float* o0 = ynext + (size_t)b * 2u * NN;
  o0[rem]      = yx / d;
  o0[NN + rem] = yy / d;
}

// ---------------------------------------------------------------------------
// Pack v (fp32 BxN) into f16 WMMA B-operand tiles (32 n x 16 b), 512 chunks.
// ---------------------------------------------------------------------------
__global__ void k_pack_v(const float* __restrict__ v, _Float16* __restrict__ vB) {
  uint32_t tid = blockIdx.x * blockDim.x + threadIdx.x;   // 2^17
  uint32_t j = tid & 7u;
  uint32_t l = (tid >> 3) & 31u;
  uint32_t c = tid >> 8;               // n-chunk < 512
  uint32_t b = l & 15u;
  uint32_t k0 = 2u * j + ((l >= 16u) ? 16u : 0u);
  _Float16* dst = vB + (size_t)c * 512u + (size_t)l * 16u + (size_t)j * 2u;
  if (b < BB) {
    size_t n = (size_t)c * 32u + k0;
    dst[0] = (_Float16)v[(size_t)b * NN + n];
    dst[1] = (_Float16)v[(size_t)b * NN + n + 1u];
  } else {
    dst[0] = (_Float16)0.0f;
    dst[1] = (_Float16)0.0f;
  }
}

// ---------------------------------------------------------------------------
// Explicitly software-pipelined WMMA accumulation over `nchunks` 1KB tile
// blocks: operands for chunks c+2/c+3 are loaded before WMMAs for c/c+1
// issue, forcing two live operand register sets (partial loadcnt waits).
// ---------------------------------------------------------------------------
__device__ __forceinline__ v8f wmma_stream(const _Float16* __restrict__ ap,
                                           const _Float16* __restrict__ bp,
                                           int nchunks) {
  v8f acc0 = {}, acc1 = {};
  v16h a0 = *(const v16h*)(ap);
  v16h b0 = *(const v16h*)(bp);
  v16h a1 = *(const v16h*)(ap + 512u);
  v16h b1 = *(const v16h*)(bp + 512u);
  for (int c = 0; c < nchunks - 2; c += 2) {
    v16h a0n = *(const v16h*)(ap + (size_t)(c + 2) * 512u);
    v16h b0n = *(const v16h*)(bp + (size_t)(c + 2) * 512u);
    v16h a1n = *(const v16h*)(ap + (size_t)(c + 3) * 512u);
    v16h b1n = *(const v16h*)(bp + (size_t)(c + 3) * 512u);
    acc0 = __builtin_amdgcn_wmma_f32_16x16x32_f16(false, a0, false, b0, (short)0, acc0, false, false);
    acc1 = __builtin_amdgcn_wmma_f32_16x16x32_f16(false, a1, false, b1, (short)0, acc1, false, false);
    a0 = a0n; b0 = b0n; a1 = a1n; b1 = b1n;
  }
  acc0 = __builtin_amdgcn_wmma_f32_16x16x32_f16(false, a0, false, b0, (short)0, acc0, false, false);
  acc1 = __builtin_amdgcn_wmma_f32_16x16x32_f16(false, a1, false, b1, (short)0, acc1, false, false);
  return acc0 + acc1;
}

// ---------------------------------------------------------------------------
// GEMM1: partial residual^T tiles.  D(16m x 16b) += A1tile(16m x 32n) * vB(32n x 16b)
// split-K = KS1 slices (32 chunks each); partials to part[s][b][m].
// ---------------------------------------------------------------------------
__global__ void k_gemm1(const _Float16* __restrict__ A1, const _Float16* __restrict__ vB,
                        float* __restrict__ part) {
  int lane = threadIdx.x & 31;
  int wave = blockIdx.x * (blockDim.x >> 5) + (threadIdx.x >> 5);  // 4096 waves
  int mt = wave >> 4;
  int s  = wave & 15;
  const _Float16* ap = A1 + ((size_t)(mt * 512 + s * 32) * 512u) + (size_t)lane * 16u;
  const _Float16* bp = vB + ((size_t)(s * 32) * 512u) + (size_t)lane * 16u;
  v8f acc = wmma_stream(ap, bp, 32);
  int b  = lane & 15;
  int hi = lane >> 4;
  if (b < BB) {
    float* outp = part + ((size_t)s * BB + b) * MM + (size_t)mt * 16u + (size_t)hi * 8u;
#pragma unroll
    for (int r = 0; r < 8; ++r) outp[r] = acc[r];
  }
}

// ---------------------------------------------------------------------------
// Pack residual = sum_s part - meas into f16 WMMA A-operand tiles (16b x 32m).
// ---------------------------------------------------------------------------
__global__ void k_pack_r(const float* __restrict__ part, const float* __restrict__ meas,
                         _Float16* __restrict__ rA) {
  uint32_t tid = blockIdx.x * blockDim.x + threadIdx.x;   // 2^15
  uint32_t j = tid & 7u;
  uint32_t l = (tid >> 3) & 31u;
  uint32_t c = tid >> 8;               // m-chunk < 128
  uint32_t b = l & 15u;
  uint32_t k0 = 2u * j + ((l >= 16u) ? 16u : 0u);
  _Float16* dst = rA + (size_t)c * 512u + (size_t)l * 16u + (size_t)j * 2u;
  if (b < BB) {
    size_t m = (size_t)c * 32u + k0;
    float e0 = -meas[(size_t)b * MM + m];
    float e1 = -meas[(size_t)b * MM + m + 1u];
#pragma unroll
    for (int s = 0; s < KS1; ++s) {
      e0 += part[(size_t)s * (BB * MM) + (size_t)b * MM + m];
      e1 += part[(size_t)s * (BB * MM) + (size_t)b * MM + m + 1u];
    }
    dst[0] = (_Float16)e0;
    dst[1] = (_Float16)e1;
  } else {
    dst[0] = (_Float16)0.0f;
    dst[1] = (_Float16)0.0f;
  }
}

// ---------------------------------------------------------------------------
// GEMM2 (split-K=2): grad_part(16b x 16n) = rA(16b x 32m) * A2tile(32m x 16n).
// ---------------------------------------------------------------------------
__global__ void k_gemm2(const _Float16* __restrict__ A2, const _Float16* __restrict__ rA,
                        float* __restrict__ gpart) {
  int lane = threadIdx.x & 31;
  int wave = blockIdx.x * (blockDim.x >> 5) + (threadIdx.x >> 5);   // 2048 waves
  int nt = wave >> 1;
  int s  = wave & 1;
  const _Float16* ap = rA + ((size_t)(s * 64) * 512u) + (size_t)lane * 16u;
  const _Float16* bp = A2 + ((size_t)(nt * 128 + s * 64) * 512u) + (size_t)lane * 16u;
  v8f acc = wmma_stream(ap, bp, 64);
  if (lane < 16) {                     // rows b = 0..7 live in lanes 0-15, VGPR r
    size_t n = (size_t)nt * 16u + lane;
    float* gp = gpart + (size_t)s * (BB * NN);
#pragma unroll
    for (int r = 0; r < 8; ++r) gp[(size_t)r * NN + n] = acc[r];
  }
}

// ---------------------------------------------------------------------------
// Finalize: x_new = v - sum_s gpart;  x = x_new;  xbar = 2*x_new - xbar.
// ---------------------------------------------------------------------------
__global__ void k_finalize(const float* __restrict__ gpart, const float* __restrict__ v,
                           float* __restrict__ x, float* __restrict__ xbar) {
  uint32_t i = blockIdx.x * blockDim.x + threadIdx.x;   // B*N = 131072
  float g = gpart[i] + gpart[(size_t)(BB * NN) + i];
  float xn = v[i] - g;
  x[i] = xn;
  xbar[i] = 2.0f * xn - xbar[i];
}

__global__ void k_copy(const float* __restrict__ x, float* __restrict__ out) {
  int i = blockIdx.x * blockDim.x + threadIdx.x;
  if (i < BB * NN) out[i] = x[i];
}

// ---------------------------------------------------------------------------
extern "C" void kernel_launch(void* const* d_in, const int* in_sizes, int n_in,
                              void* d_out, int out_size, void* d_ws, size_t ws_size,
                              hipStream_t stream) {
  const float* meas = (const float*)d_in[0];   // (8, 4096)
  const float* A    = (const float*)d_in[1];   // (4096, 16384)

  char* ws = (char*)d_ws;
  size_t off = 0;
  _Float16* A1 = (_Float16*)(ws + off); off += (size_t)256 * 512 * 1024;       // 128 MiB
  _Float16* A2 = (_Float16*)(ws + off); off += (size_t)1024 * 128 * 1024;      // 128 MiB
  float* x    = (float*)(ws + off); off += (size_t)BB * NN * 4;                // 512 KiB
  float* xbar = (float*)(ws + off); off += (size_t)BB * NN * 4;                // 512 KiB
  float* yA   = (float*)(ws + off); off += (size_t)BB * 2 * NN * 4;            // 1 MiB
  float* yB   = (float*)(ws + off); off += (size_t)BB * 2 * NN * 4;            // 1 MiB
  float* v    = (float*)(ws + off); off += (size_t)BB * NN * 4;                // 512 KiB
  _Float16* vB = (_Float16*)(ws + off); off += (size_t)512 * 1024;             // 512 KiB
  float* part  = (float*)(ws + off); off += (size_t)KS1 * BB * MM * 4;         // 2 MiB
  _Float16* rA = (_Float16*)(ws + off); off += (size_t)128 * 1024;             // 128 KiB
  float* gpart = (float*)(ws + off); off += (size_t)KS2 * BB * NN * 4;         // 1 MiB

  // one-time (per call) A conversions into operand-ready f16 tiles
  k_conv_a1<<<131072, 256, 0, stream>>>(A, A1);
  k_conv_a2<<<131072, 256, 0, stream>>>(A, A2);
  // zero x, xbar, yA, yB (laid out contiguously)
  k_zero<<<3072, 256, 0, stream>>>(x, BB * NN * 2 + BB * 2 * NN * 2);

  float* ycur = yA;
  float* ynext = yB;
  for (int it = 0; it < NITER; ++it) {
    k_pointwise<<<512, 256, 0, stream>>>(x, xbar, ycur, ynext, v);
    k_pack_v  <<<512, 256, 0, stream>>>(v, vB);
    k_gemm1   <<<512, 256, 0, stream>>>(A1, vB, part);
    k_pack_r  <<<128, 256, 0, stream>>>(part, meas, rA);
    k_gemm2   <<<256, 256, 0, stream>>>(A2, rA, gpart);
    k_finalize<<<512, 256, 0, stream>>>(gpart, v, x, xbar);
    float* t = ycur; ycur = ynext; ynext = t;
  }
  k_copy<<<512, 256, 0, stream>>>(x, (float*)d_out);
}
